// MultiHeadedAttention_884763263505
// MI455X (gfx1250) — compile-verified, matrix operand order checked by reading, not run on hardware
//
#include <hip/hip_runtime.h>
#include <hip/hip_bf16.h>
#include <stdint.h>

// ---------------------------------------------------------------------------
// MI455X (gfx1250) multi-head attention
//   - all GEMMs via v_wmma_f32_16x16x32_f16 (f16 in, f32 accumulate)
//   - K/V tiles staged to LDS with global_load_async_to_lds_b128 (ASYNCcnt)
//   - 16x64 register blocking per wave in the dense GEMMs
// ---------------------------------------------------------------------------
typedef __attribute__((ext_vector_type(16))) _Float16 v16h;
typedef __attribute__((ext_vector_type(8)))  _Float16 v8h;
typedef __attribute__((ext_vector_type(4)))  _Float16 v4h;
typedef __attribute__((ext_vector_type(8)))  float    v8f;

#define DEVFN static __device__ __forceinline__

constexpr int Bc  = 2;
constexpr int Ac  = 6;
constexpr int Sc  = 1024;
constexpr int Hc  = 8;
constexpr int DKc = 64;
constexpr int INc = 256;   // q/k/v feature dim
constexpr int DMc = 512;   // d_model
constexpr int ROWS = Bc * Ac * Sc;  // 12288 token rows

// --- WMMA wrapper (8-arg form, probe-verified signature) -------------------
DEVFN v8f wmma(v16h a, v16h b, v8f c) {
  return __builtin_amdgcn_wmma_f32_16x16x32_f16(
      /*neg_a=*/false, a, /*neg_b=*/false, b,
      /*c_mod=*/(short)0, c, /*reuse_a=*/false, /*reuse_b=*/false);
}

// --- async global -> LDS staging (ASYNCcnt path) ---------------------------
DEVFN void async_ld_b128(uint32_t lds_off, const void* gaddr) {
  asm volatile("global_load_async_to_lds_b128 %0, %1, off"
               :: "v"(lds_off), "v"(gaddr) : "memory");
}
DEVFN void wait_async0() { asm volatile("s_wait_asynccnt 0" ::: "memory"); }
DEVFN void wait_ds0()    { asm volatile("s_wait_dscnt 0"    ::: "memory"); }
DEVFN uint32_t lds_off(const void* p) { return (uint32_t)(uintptr_t)p; }

// --- A/B fragment gather, 16-bit 16x32 layout (ISA 7.12.2) -----------------
// lane L: row (or B-column) = L%16; packed halves:
//   elems 0..7  -> K = k0 + (L>=16 ? 8 : 0) + i        (contiguous, 16 B)
//   elems 8..15 -> K = k0 + 16 + (L>=16 ? 8 : 0) + i   (contiguous, 16 B)
DEVFN v16h frag_f16(const _Float16* __restrict__ base, int ld, int rbase, int k0) {
  const int lane = threadIdx.x & 31;
  const int row  = rbase + (lane & 15);
  const int hi   = (lane >> 4) & 1;
  const _Float16* p = base + (size_t)row * ld + k0 + hi * 8;
  v8h lo = *(const v8h*)(p);        // K = k0 + hi*8 .. +7
  v8h hv = *(const v8h*)(p + 16);   // K = k0 + 16 + hi*8 .. +7
  v16h r;
#pragma unroll
  for (int i = 0; i < 8; ++i) { r[i] = lo[i]; r[i + 8] = hv[i]; }
  return r;
}

// --- Kernel 0a: weight convert + transpose (fp32 [K,N] -> f16 [N,K]) -------
__global__ void wconvT(const float* __restrict__ in, _Float16* __restrict__ out,
                       int K, int N) {
  int i = blockIdx.x * blockDim.x + threadIdx.x;
  if (i >= K * N) return;
  int k = i / N, n = i % N;
  out[(size_t)n * K + k] = (_Float16)in[i];
}

// --- Kernel 0b: activation convert fp32 -> f16 (same layout, vectorized) ---
__global__ void cvt16(const float* __restrict__ in, _Float16* __restrict__ out,
                      int n4) {
  int i = blockIdx.x * blockDim.x + threadIdx.x;
  if (i >= n4) return;
  float4 v = ((const float4*)in)[i];
  v4h h; h[0] = (_Float16)v.x; h[1] = (_Float16)v.y;
         h[2] = (_Float16)v.z; h[3] = (_Float16)v.w;
  ((v4h*)out)[i] = h;
}

// --- Kernel 1: QKV projection GEMM (M=12288, K=256, N=512), 16x64 per wave -
// out layout: transposed==0 -> [BA,H,S,DK] (Q,K);  transposed==1 -> [BA,H,DK,S] (V^T)
__global__ void proj_gemm(const _Float16* __restrict__ X,   // [12288,256] f16
                          const _Float16* __restrict__ WT,  // [512,256] f16
                          const float* __restrict__ bias,   // [512] f32
                          _Float16* __restrict__ out, int transposed) {
  const int lane    = threadIdx.x & 31;
  const int colbase = (blockIdx.x * blockDim.y + threadIdx.y) * 64;  // 0..448
  const int mbase   = blockIdx.y * 16;
  v8f acc0 = {}, acc1 = {}, acc2 = {}, acc3 = {};
#pragma unroll
  for (int k0 = 0; k0 < INc; k0 += 32) {
    v16h a  = frag_f16(X,  INc, mbase, k0);            // reused 4x
    v16h b0 = frag_f16(WT, INc, colbase,      k0);
    v16h b1 = frag_f16(WT, INc, colbase + 16, k0);
    v16h b2 = frag_f16(WT, INc, colbase + 32, k0);
    v16h b3 = frag_f16(WT, INc, colbase + 48, k0);
    acc0 = wmma(a, b0, acc0);
    acc1 = wmma(a, b1, acc1);
    acc2 = wmma(a, b2, acc2);
    acc3 = wmma(a, b3, acc3);
  }
  const int g = lane >> 4, n = lane & 15;
  const int h = colbase >> 6;  // 64-aligned group => single head
  v8f* accs[4] = {&acc0, &acc1, &acc2, &acc3};
#pragma unroll
  for (int j = 0; j < 4; ++j) {
    const int col = colbase + j * 16 + n;
    const int d   = col & 63;
    const float bc = bias[col];
#pragma unroll
    for (int r = 0; r < 8; ++r) {
      int gr = mbase + r + 8 * g;            // global token row
      int ba = gr >> 10, s = gr & (Sc - 1);
      float v = (*accs[j])[r] + bc;
      size_t idx = transposed
          ? ((size_t)((ba * Hc + h) * DKc + d)) * Sc + s
          : ((size_t)((ba * Hc + h) * Sc + s)) * DKc + d;
      out[idx] = (_Float16)v;
    }
  }
}

// --- Kernel 2: flash attention -----------------------------------------------
// block = 4 waves, one head; each wave owns 16 query rows. K/V key-blocks of 32
// are staged once per block into LDS via async loads, double buffered.
__global__ void flash_attn(const _Float16* __restrict__ Q,   // [BAH,S,DK]
                           const _Float16* __restrict__ K,   // [BAH,S,DK]
                           const _Float16* __restrict__ Vt,  // [BAH,DK,S]
                           _Float16* __restrict__ Oc) {      // [BA,S,512]
  __shared__ __align__(16) _Float16 Kl[2][32 * 64];  // [key 32][dk 64]
  __shared__ __align__(16) _Float16 Vl[2][64 * 32];  // [dk 64][key 32]
  __shared__ __align__(16) _Float16 Pl[4][16 * 32];  // per-wave P tile

  const int lane = threadIdx.x & 31;
  const int wv   = threadIdx.y;
  const int tid  = wv * 32 + lane;           // 0..127
  const int bah  = blockIdx.y;               // 0..95
  const int ba   = bah / Hc, h = bah % Hc;
  const _Float16* Qh = Q  + (size_t)bah * Sc * DKc;
  const _Float16* Kh = K  + (size_t)bah * Sc * DKc;
  const _Float16* Vh = Vt + (size_t)bah * DKc * Sc;
  const int qbase = (blockIdx.x * 4 + wv) * 16;

  // stage one 32-key block of K and V^T into LDS buffer `buf`
  auto stage = [&](int kb, int buf) {
    const char* kg = (const char*)(Kh + (size_t)kb * DKc);  // 4 KB contiguous
#pragma unroll
    for (int t = 0; t < 2; ++t) {
      int c = tid + t * 128;  // 16-byte chunk id, 0..255
      async_ld_b128(lds_off((const char*)&Kl[buf][0] + c * 16), kg + c * 16);
    }
#pragma unroll
    for (int t = 0; t < 2; ++t) {
      int c = tid + t * 128;
      int d = c >> 2, seg = c & 3;           // 64 rows x 4 chunks
      async_ld_b128(lds_off((const char*)&Vl[buf][0] + d * 64 + seg * 16),
                    (const char*)(Vh + (size_t)d * Sc + kb) + seg * 16);
    }
  };

  // Q fragments, reused across all key blocks (K = 64 -> two 16x32 frags)
  v16h qa0 = frag_f16(Qh, DKc, qbase, 0);
  v16h qa1 = frag_f16(Qh, DKc, qbase, 32);

  v8f o0 = {}, o1 = {}, o2 = {}, o3 = {};
  float m[8], l[8];
#pragma unroll
  for (int r = 0; r < 8; ++r) { m[r] = -1e30f; l[r] = 0.f; }
  const float scale = 0.125f;  // 1/sqrt(64)
  _Float16* P = &Pl[wv][0];
  const int g = lane >> 4, n = lane & 15;

  stage(0, 0);
  wait_async0();
  __syncthreads();

  constexpr int NB = Sc / 32;  // 32 key blocks
  for (int i = 0; i < NB; ++i) {
    const int cur = i & 1;
    if (i + 1 < NB) stage((i + 1) * 32, cur ^ 1);   // prefetch next block

    const _Float16* Kt = &Kl[cur][0];
    const _Float16* Vb = &Vl[cur][0];

    // scores S = Q(16x64) * K^T(64x32) -> two 16x16 C frags, 4 WMMAs
    v8f s0 = {}, s1 = {};
    {
      v16h b0 = frag_f16(Kt, DKc, 0,  0);
      v16h b1 = frag_f16(Kt, DKc, 0,  32);
      v16h b2 = frag_f16(Kt, DKc, 16, 0);
      v16h b3 = frag_f16(Kt, DKc, 16, 32);
      s0 = wmma(qa0, b0, s0);  s0 = wmma(qa1, b1, s0);
      s1 = wmma(qa0, b2, s1);  s1 = wmma(qa1, b3, s1);
    }

    // online softmax on C-fragment layout (rows r+8g, cols across 16 lanes)
    float p0[8], p1[8], alpha[8];
#pragma unroll
    for (int r = 0; r < 8; ++r) {
      float a0 = s0[r] * scale, a1 = s1[r] * scale;
      float bm = fmaxf(a0, a1);
#pragma unroll
      for (int off = 1; off < 16; off <<= 1)
        bm = fmaxf(bm, __shfl_xor(bm, off, 32));
      float mn = fmaxf(m[r], bm);
      alpha[r] = __expf(m[r] - mn);
      p0[r] = __expf(a0 - mn);
      p1[r] = __expf(a1 - mn);
      float rs = p0[r] + p1[r];
#pragma unroll
      for (int off = 1; off < 16; off <<= 1)
        rs += __shfl_xor(rs, off, 32);
      l[r] = l[r] * alpha[r] + rs;
      m[r] = mn;
    }
#pragma unroll
    for (int r = 0; r < 8; ++r) {
      o0[r] *= alpha[r]; o1[r] *= alpha[r];
      o2[r] *= alpha[r]; o3[r] *= alpha[r];
    }

    // C-layout -> row-major 16x32 P tile in LDS (per-wave slice)
#pragma unroll
    for (int r = 0; r < 8; ++r) {
      int row = r + 8 * g;
      P[row * 32 + n]      = (_Float16)p0[r];
      P[row * 32 + 16 + n] = (_Float16)p1[r];
    }
    wait_ds0();  // order DS store -> DS load within the wave

    v16h pa = frag_f16(P, 32, 0, 0);  // reload P as an A fragment

    // O += P(16x32) * V(32x64); B frags contiguous thanks to V^T LDS layout
    v16h vb0 = frag_f16(Vb, 32, 0,  0);
    v16h vb1 = frag_f16(Vb, 32, 16, 0);
    v16h vb2 = frag_f16(Vb, 32, 32, 0);
    v16h vb3 = frag_f16(Vb, 32, 48, 0);
    o0 = wmma(pa, vb0, o0);
    o1 = wmma(pa, vb1, o1);
    o2 = wmma(pa, vb2, o2);
    o3 = wmma(pa, vb3, o3);

    wait_async0();     // next block's K/V landed in LDS (per-wave counter)
    __syncthreads();   // all waves done with buffer `cur` + see new data
  }

  // normalize + write head-concatenated f16 activation [BA,S,512]
#pragma unroll
  for (int r = 0; r < 8; ++r) {
    float inv = 1.0f / l[r];
    int s = qbase + r + 8 * g;
    _Float16* dst = Oc + ((size_t)(ba * Sc + s)) * DMc + h * DKc;
    dst[n]      = (_Float16)(o0[r] * inv);
    dst[16 + n] = (_Float16)(o1[r] * inv);
    dst[32 + n] = (_Float16)(o2[r] * inv);
    dst[48 + n] = (_Float16)(o3[r] * inv);
  }
}

// --- Kernel 3: output projection (M=12288, K=512, N=256), 16x64 per wave ---
__global__ void out_proj(const _Float16* __restrict__ Oc,    // [12288,512] f16
                         const _Float16* __restrict__ WoT,   // [256,512] f16
                         const float* __restrict__ bo,       // [256] f32
                         float* __restrict__ out) {          // [12288,256] f32
  const int lane    = threadIdx.x & 31;
  const int colbase = (blockIdx.x * blockDim.y + threadIdx.y) * 64;  // 0..192
  const int mbase   = blockIdx.y * 16;
  v8f acc0 = {}, acc1 = {}, acc2 = {}, acc3 = {};
#pragma unroll
  for (int k0 = 0; k0 < DMc; k0 += 32) {
    v16h a  = frag_f16(Oc,  DMc, mbase, k0);           // reused 4x
    v16h b0 = frag_f16(WoT, DMc, colbase,      k0);
    v16h b1 = frag_f16(WoT, DMc, colbase + 16, k0);
    v16h b2 = frag_f16(WoT, DMc, colbase + 32, k0);
    v16h b3 = frag_f16(WoT, DMc, colbase + 48, k0);
    acc0 = wmma(a, b0, acc0);
    acc1 = wmma(a, b1, acc1);
    acc2 = wmma(a, b2, acc2);
    acc3 = wmma(a, b3, acc3);
  }
  const int g = lane >> 4, n = lane & 15;
  v8f* accs[4] = {&acc0, &acc1, &acc2, &acc3};
#pragma unroll
  for (int j = 0; j < 4; ++j) {
    const int col = colbase + j * 16 + n;
    const float bc = bo[col];
#pragma unroll
    for (int r = 0; r < 8; ++r) {
      int gr = mbase + r + 8 * g;
      out[(size_t)gr * INc + col] = (*accs[j])[r] + bc;
    }
  }
}

// ---------------------------------------------------------------------------
extern "C" void kernel_launch(void* const* d_in, const int* in_sizes, int n_in,
                              void* d_out, int out_size, void* d_ws, size_t ws_size,
                              hipStream_t stream) {
  (void)in_sizes; (void)n_in; (void)out_size;
  const float* q  = (const float*)d_in[0];
  const float* k  = (const float*)d_in[1];
  const float* v  = (const float*)d_in[2];
  const float* Wq = (const float*)d_in[3];
  const float* bq = (const float*)d_in[4];
  const float* Wk = (const float*)d_in[5];
  const float* bk = (const float*)d_in[6];
  const float* Wv = (const float*)d_in[7];
  const float* bv = (const float*)d_in[8];
  const float* Wo = (const float*)d_in[9];
  const float* bo = (const float*)d_in[10];
  float* out = (float*)d_out;

  // workspace carve-up (f16 elements)
  _Float16* ws = (_Float16*)d_ws;
  size_t off = 0;
  auto take = [&](size_t nelem) { _Float16* p = ws + off; off += nelem; return p; };
  _Float16* WqT = take((size_t)DMc * INc);          // [512,256]
  _Float16* WkT = take((size_t)DMc * INc);
  _Float16* WvT = take((size_t)DMc * INc);
  _Float16* WoT = take((size_t)INc * DMc);          // [256,512]
  _Float16* Xq  = take((size_t)ROWS * INc);         // f16 activations
  _Float16* Xk  = take((size_t)ROWS * INc);
  _Float16* Xv  = take((size_t)ROWS * INc);
  _Float16* Qh  = take((size_t)Bc * Ac * Hc * Sc * DKc);
  _Float16* Kh  = take((size_t)Bc * Ac * Hc * Sc * DKc);
  _Float16* Vt  = take((size_t)Bc * Ac * Hc * DKc * Sc);
  _Float16* Oc  = take((size_t)ROWS * DMc);
  if (off * sizeof(_Float16) > ws_size) return;     // scratch insufficient

  // 0) convert weights (transposed) and activations to f16
  int wn = INc * DMc;
  wconvT<<<(wn + 255) / 256, 256, 0, stream>>>(Wq, WqT, INc, DMc);
  wconvT<<<(wn + 255) / 256, 256, 0, stream>>>(Wk, WkT, INc, DMc);
  wconvT<<<(wn + 255) / 256, 256, 0, stream>>>(Wv, WvT, DMc, INc == INc ? INc : INc);  // (K=DMc,N=INc)
  wconvT<<<(wn + 255) / 256, 256, 0, stream>>>(Wo, WoT, DMc, INc);
  int n4 = ROWS * INc / 4;
  cvt16<<<(n4 + 255) / 256, 256, 0, stream>>>(q, Xq, n4);
  cvt16<<<(n4 + 255) / 256, 256, 0, stream>>>(k, Xk, n4);
  cvt16<<<(n4 + 255) / 256, 256, 0, stream>>>(v, Xv, n4);

  dim3 blk(32, 4);  // 4 wave32s per block
  // 1) QKV projections: grid (512/64/4, 12288/16)
  proj_gemm<<<dim3(DMc / 64 / 4, ROWS / 16), blk, 0, stream>>>(Xq, WqT, bq, Qh, 0);
  proj_gemm<<<dim3(DMc / 64 / 4, ROWS / 16), blk, 0, stream>>>(Xk, WkT, bk, Kh, 0);
  proj_gemm<<<dim3(DMc / 64 / 4, ROWS / 16), blk, 0, stream>>>(Xv, WvT, bv, Vt, 1);
  // 2) attention: grid (1024/16/4, 96)
  flash_attn<<<dim3(Sc / 16 / 4, Bc * Ac * Hc), blk, 0, stream>>>(Qh, Kh, Vt, Oc);
  // 3) output projection: grid (256/64/4, 12288/16)
  out_proj<<<dim3(INc / 64 / 4, ROWS / 16), blk, 0, stream>>>(Oc, WoT, bo, out);
}
